// GAT_29265907155155
// MI455X (gfx1250) — matrix operand order repaired and checked by reading, not compiled, not run on hardware
//
#include <hip/hip_runtime.h>
#include <hip/hip_bf16.h>
#include <math.h>

// ---------------------------------------------------------------------------
// GAT on MI455X (gfx1250). Dense projections via v_wmma_f32_16x16x32_bf16
// (64-row macro-tiles, B-fragment reused across 4 WMMAs), edge softmax via
// max / exp-sum / alpha / vectorized scatter passes (L2-resident
// accumulators), LayerNorm via wave32 shuffle reductions.
// ---------------------------------------------------------------------------

#define N_HID 128
#define HEADS 4
#define CHAN 32
#define NEG_SLOPE 0.2f
#define LN_EPS 1e-5f

typedef __attribute__((ext_vector_type(16))) __bf16 v16bf;
typedef __attribute__((ext_vector_type(8)))  float  v8f;

// ---------------- ordered-float encoding for atomic max --------------------
__device__ __forceinline__ unsigned ordf(float f) {
    unsigned u = __float_as_uint(f);
    return (u & 0x80000000u) ? ~u : (u | 0x80000000u);
}
__device__ __forceinline__ float deordf(unsigned k) {
    return (k & 0x80000000u) ? __uint_as_float(k ^ 0x80000000u)
                             : __uint_as_float(~k);
}
__device__ __forceinline__ float lrelu(float x) {
    return x > 0.f ? x : NEG_SLOPE * x;
}

// ---------------- WMMA GEMM: Y[N,128] = X[N,K] @ W[K,128] (+bias) ----------
// Block = 256 threads = 8 waves. Block covers 64 rows x 128 cols; wave w
// owns columns [16w,16w+16) and all four 16-row tiles, so each B fragment
// feeds 4 v_wmma ops. K in {64,128}, stepped 32 at a time.
template <int K, bool HAS_BIAS>
__global__ void gemm_wmma_bf16(const float* __restrict__ X,
                               const float* __restrict__ W,
                               const float* __restrict__ bias,
                               float* __restrict__ Y, int N) {
    __shared__ float Xs[64 * K];

    const int tid  = threadIdx.x;
    const int wave = tid >> 5;
    const int lane = tid & 31;
    const int hi   = lane >> 4;     // 0: lanes 0-15, 1: lanes 16-31
    const int lo   = lane & 15;
    const int m0   = blockIdx.x * 64;
    const int col0 = wave * 16;
    const int col  = col0 + lo;

    // stage 64xK fp32 rows of X in LDS (zero-pad tail rows)
    for (int idx = tid; idx < 64 * K; idx += 256) {
        const int r = idx / K, k = idx % K;
        const int row = m0 + r;
        Xs[idx] = (row < N) ? X[(size_t)row * K + k] : 0.0f;
    }
    __syncthreads();

    v8f acc0 = {}, acc1 = {}, acc2 = {}, acc3 = {};
    for (int k0 = 0; k0 < K; k0 += 32) {
        // ---- B fragment: 32x16 bf16; K rows 0-15 -> lanes 0-15, +16 -> hi
        v16bf b;
        #pragma unroll
        for (int v = 0; v < 8; ++v) {
            const int kr = k0 + 2 * v + 16 * hi;
            b[2 * v]     = (__bf16)W[(size_t)kr * N_HID + col];
            b[2 * v + 1] = (__bf16)W[(size_t)(kr + 1) * N_HID + col];
        }
        // ---- A fragments: 16x32 bf16 per row-tile, k = 16*hi + element, same order as B
        #pragma unroll
        for (int t = 0; t < 4; ++t) {
            v16bf a;
            #pragma unroll
            for (int v = 0; v < 8; ++v) {
                const int kk = k0 + 2 * v + 16 * hi;
                a[2 * v]     = (__bf16)Xs[(16 * t + lo) * K + kk];
                a[2 * v + 1] = (__bf16)Xs[(16 * t + lo) * K + kk + 1];
            }
            if (t == 0) acc0 = __builtin_amdgcn_wmma_f32_16x16x32_bf16(false, a, false, b, (short)0, acc0, false, false);
            if (t == 1) acc1 = __builtin_amdgcn_wmma_f32_16x16x32_bf16(false, a, false, b, (short)0, acc1, false, false);
            if (t == 2) acc2 = __builtin_amdgcn_wmma_f32_16x16x32_bf16(false, a, false, b, (short)0, acc2, false, false);
            if (t == 3) acc3 = __builtin_amdgcn_wmma_f32_16x16x32_bf16(false, a, false, b, (short)0, acc3, false, false);
        }
    }

    const float bb = HAS_BIAS ? bias[col] : 0.0f;   // invariant per lane

    // ---- store D: tile t, VGPR v -> row m0 + 16t + v + 8*hi, col ---------
    if (m0 + 64 <= N) {
        #pragma unroll
        for (int t = 0; t < 4; ++t) {
            const v8f acc = (t == 0) ? acc0 : (t == 1) ? acc1 : (t == 2) ? acc2 : acc3;
            #pragma unroll
            for (int v = 0; v < 8; ++v) {
                const int row = m0 + 16 * t + v + 8 * hi;
                Y[(size_t)row * N_HID + col] = acc[v] + bb;
            }
        }
    } else {
        #pragma unroll
        for (int t = 0; t < 4; ++t) {
            const v8f acc = (t == 0) ? acc0 : (t == 1) ? acc1 : (t == 2) ? acc2 : acc3;
            #pragma unroll
            for (int v = 0; v < 8; ++v) {
                const int row = m0 + 16 * t + v + 8 * hi;
                if (row < N) Y[(size_t)row * N_HID + col] = acc[v] + bb;
            }
        }
    }
}

// ---------------- attention coefficients: al = einsum(nhc,hc->nh) ----------
__global__ void attn_coef_kernel(const float* __restrict__ XL,
                                 const float* __restrict__ a_src,
                                 const float* __restrict__ a_dst,
                                 float* __restrict__ SA,
                                 float* __restrict__ SD,
                                 int N, int H, int C) {
    const int idx = blockIdx.x * blockDim.x + threadIdx.x;
    if (idx >= N * H) return;
    const int n = idx / H, h = idx - n * H;
    const float* row = XL + (size_t)n * H * C + h * C;
    const float* as = a_src + h * C;
    const float* ad = a_dst + h * C;
    float ss = 0.f, sd = 0.f;
    for (int c = 0; c < C; ++c) {
        const float v = row[c];
        ss += v * as[c];
        sd += v * ad[c];
    }
    SA[idx] = ss;
    SD[idx] = sd;
}

// ---------------- pass 1: per-destination segment max (thread per edge) ----
__global__ void edge_max_kernel(const int* __restrict__ src,
                                const int* __restrict__ dst,
                                const float* __restrict__ SA,
                                const float* __restrict__ SD,
                                unsigned* __restrict__ M,
                                int E, int N, int H) {
    const int e = blockIdx.x * blockDim.x + threadIdx.x;
    if (e >= E + N) return;                     // +N self loops
    int s, d;
    if (e < E) { s = src[e]; d = dst[e]; } else { s = d = e - E; }
    for (int h = 0; h < H; ++h) {
        const float sc = lrelu(SA[s * H + h] + SD[d * H + h]);
        atomicMax(&M[d * H + h], ordf(sc));
    }
}

// ---------------- pass 2: exp-sum (softmax denominator) --------------------
__global__ void edge_expsum_kernel(const int* __restrict__ src,
                                   const int* __restrict__ dst,
                                   const float* __restrict__ SA,
                                   const float* __restrict__ SD,
                                   const unsigned* __restrict__ M,
                                   float* __restrict__ DEN,
                                   int E, int N, int H) {
    const int e = blockIdx.x * blockDim.x + threadIdx.x;
    if (e >= E + N) return;
    int s, d;
    if (e < E) { s = src[e]; d = dst[e]; } else { s = d = e - E; }
    for (int h = 0; h < H; ++h) {
        const float sc = lrelu(SA[s * H + h] + SD[d * H + h]);
        const float m = deordf(M[d * H + h]);
        atomicAdd(&DEN[d * H + h], expf(sc - m));
    }
}

// ---------------- pass 3: per-edge attention weights -----------------------
__global__ void edge_alpha_kernel(const int* __restrict__ src,
                                  const int* __restrict__ dst,
                                  const float* __restrict__ SA,
                                  const float* __restrict__ SD,
                                  const unsigned* __restrict__ M,
                                  const float* __restrict__ DEN,
                                  float* __restrict__ ALPHA,
                                  int E, int N, int H) {
    const int e = blockIdx.x * blockDim.x + threadIdx.x;
    if (e >= E + N) return;
    int s, d;
    if (e < E) { s = src[e]; d = dst[e]; } else { s = d = e - E; }
    for (int h = 0; h < H; ++h) {
        const float sc = lrelu(SA[s * H + h] + SD[d * H + h]);
        const float m   = deordf(M[d * H + h]);
        const float den = DEN[d * H + h];
        ALPHA[e * H + h] = expf(sc - m) / (den + 1e-16f);
    }
}

// ---------------- pass 4: alpha-weighted scatter, HC=128, float4 -----------
__global__ void edge_accum128_kernel(const int* __restrict__ src,
                                     const int* __restrict__ dst,
                                     const float* __restrict__ ALPHA,
                                     const float* __restrict__ XL,
                                     float* __restrict__ ACC,
                                     int E, int N) {
    const long long idx = (long long)blockIdx.x * blockDim.x + threadIdx.x;
    const int ET = E + N;
    if (idx >= (long long)ET * 32) return;
    const int e = (int)(idx >> 5);
    const int q = (int)(idx & 31);
    const int j = q << 2;                       // channel group of 4
    const int h = q >> 3;                       // head = j / 32
    int s, d;
    if (e < E) { s = src[e]; d = dst[e]; } else { s = d = e - E; }
    const float a = ALPHA[e * HEADS + h];
    const float4 xv = *(const float4*)(XL + (size_t)s * N_HID + j);
    float* acc = ACC + (size_t)d * N_HID + j;
    atomicAdd(acc + 0, a * xv.x);
    atomicAdd(acc + 1, a * xv.y);
    atomicAdd(acc + 2, a * xv.z);
    atomicAdd(acc + 3, a * xv.w);
}

// ---------------- pass 4 (output layer): HC=2, thread per edge -------------
__global__ void edge_accum2_kernel(const int* __restrict__ src,
                                   const int* __restrict__ dst,
                                   const float* __restrict__ ALPHA,
                                   const float* __restrict__ XL,
                                   float* __restrict__ ACC,
                                   int E, int N) {
    const int e = blockIdx.x * blockDim.x + threadIdx.x;
    if (e >= E + N) return;
    int s, d;
    if (e < E) { s = src[e]; d = dst[e]; } else { s = d = e - E; }
    const float a = ALPHA[e];
    atomicAdd(&ACC[d * 2 + 0], a * XL[s * 2 + 0]);
    atomicAdd(&ACC[d * 2 + 1], a * XL[s * 2 + 1]);
}

// ---------------- finalize: h = relu(LN(acc + bias)) + h (in place) --------
// One wave32 per node; lane handles channels {lane, lane+32, lane+64, lane+96}
__global__ void finalize_ln_kernel(const float* __restrict__ ACC,
                                   const float* __restrict__ bias,
                                   const float* __restrict__ g,
                                   const float* __restrict__ be,
                                   float* __restrict__ Hio, int N) {
    const int wave = threadIdx.x >> 5;
    const int lane = threadIdx.x & 31;
    const int n = blockIdx.x * 8 + wave;
    if (n >= N) return;

    float v[4];
    float s = 0.f, s2 = 0.f;
    #pragma unroll
    for (int k = 0; k < 4; ++k) {
        const int c = lane + 32 * k;
        v[k] = ACC[(size_t)n * N_HID + c] + bias[c];
        s  += v[k];
        s2 += v[k] * v[k];
    }
    #pragma unroll
    for (int m = 16; m >= 1; m >>= 1) {
        s  += __shfl_xor(s,  m, 32);
        s2 += __shfl_xor(s2, m, 32);
    }
    const float mean = s * (1.0f / N_HID);
    const float var  = s2 * (1.0f / N_HID) - mean * mean;
    const float rstd = rsqrtf(var + LN_EPS);
    #pragma unroll
    for (int k = 0; k < 4; ++k) {
        const int c = lane + 32 * k;
        float y = (v[k] - mean) * rstd * g[c] + be[c];
        y = fmaxf(y, 0.0f);
        Hio[(size_t)n * N_HID + c] += y;        // residual add in place
    }
}

// ---------------- output projection + output attention coefficients --------
__global__ void project_out_kernel(const float* __restrict__ Hin,
                                   const float* __restrict__ Wo,
                                   const float* __restrict__ aso,
                                   const float* __restrict__ ado,
                                   float* __restrict__ XLO,
                                   float* __restrict__ SA,
                                   float* __restrict__ SD, int N) {
    const int wave = threadIdx.x >> 5;
    const int lane = threadIdx.x & 31;
    const int n = blockIdx.x * 8 + wave;
    if (n >= N) return;
    float p0 = 0.f, p1 = 0.f;
    #pragma unroll
    for (int k = 0; k < 4; ++k) {
        const int c = lane + 32 * k;
        const float hv = Hin[(size_t)n * N_HID + c];
        p0 += hv * Wo[c * 2 + 0];
        p1 += hv * Wo[c * 2 + 1];
    }
    #pragma unroll
    for (int m = 16; m >= 1; m >>= 1) {
        p0 += __shfl_xor(p0, m, 32);
        p1 += __shfl_xor(p1, m, 32);
    }
    if (lane == 0) {
        XLO[n * 2 + 0] = p0;
        XLO[n * 2 + 1] = p1;
        SA[n] = p0 * aso[0] + p1 * aso[1];
        SD[n] = p0 * ado[0] + p1 * ado[1];
    }
}

__global__ void finalize_out_kernel(const float* __restrict__ ACC,
                                    const float* __restrict__ bo,
                                    float* __restrict__ out, int N) {
    const int idx = blockIdx.x * blockDim.x + threadIdx.x;
    if (idx >= N * 2) return;
    out[idx] = ACC[idx] + bo[idx & 1];
}

// ---------------------------------------------------------------------------
static inline int cdiv(long long a, long long b) { return (int)((a + b - 1) / b); }

extern "C" void kernel_launch(void* const* d_in, const int* in_sizes, int n_in,
                              void* d_out, int out_size, void* d_ws, size_t ws_size,
                              hipStream_t stream) {
    // inputs in setup_inputs() order
    const float* x   = (const float*)d_in[0];
    const int*   ei  = (const int*)  d_in[1];
    const float* W1  = (const float*)d_in[2];
    const float* as1 = (const float*)d_in[3];
    const float* ad1 = (const float*)d_in[4];
    const float* b1  = (const float*)d_in[5];
    const float* g1  = (const float*)d_in[6];
    const float* be1 = (const float*)d_in[7];
    const float* rw  = (const float*)d_in[8];
    const float* rb  = (const float*)d_in[9];
    const float* W2  = (const float*)d_in[10];
    const float* as2 = (const float*)d_in[11];
    const float* ad2 = (const float*)d_in[12];
    const float* b2  = (const float*)d_in[13];
    const float* g2  = (const float*)d_in[14];
    const float* be2 = (const float*)d_in[15];
    const float* W3  = (const float*)d_in[16];
    const float* as3 = (const float*)d_in[17];
    const float* ad3 = (const float*)d_in[18];
    const float* b3  = (const float*)d_in[19];
    const float* g3  = (const float*)d_in[20];
    const float* be3 = (const float*)d_in[21];
    const float* Wo  = (const float*)d_in[22];
    const float* aso = (const float*)d_in[23];
    const float* ado = (const float*)d_in[24];
    const float* bo  = (const float*)d_in[25];

    const int N = in_sizes[0] / 64;       // 50000
    const int E = in_sizes[1] / 2;        // 800000
    const int ET = E + N;                 // edges + self loops
    const int* src = ei;
    const int* dst = ei + E;

    // workspace (fp32): XL | ACC | H | SA | SD | M(u32) | DEN | ALPHA
    float*  ws  = (float*)d_ws;
    const size_t nf = (size_t)N * N_HID;
    float*    XL    = ws;
    float*    ACC   = ws + nf;
    float*    Hb    = ws + 2 * nf;
    float*    SA    = ws + 3 * nf;
    float*    SD    = SA + (size_t)N * HEADS;
    unsigned* Mx    = (unsigned*)(SD + (size_t)N * HEADS);
    float*    DEN   = (float*)(Mx + (size_t)N * HEADS);
    float*    ALPHA = DEN + (size_t)N * HEADS;

    const int mtiles = cdiv(N, 64);
    const int egrid  = cdiv(ET, 256);

    // residual projection straight into H: H = x @ rw + rb
    gemm_wmma_bf16<64, true><<<mtiles, 256, 0, stream>>>(x, rw, rb, Hb, N);

    // ---------------- three hidden GAT layers ----------------
    const float* layer_in[3] = {x, Hb, Hb};
    const float* Ws[3]  = {W1, W2, W3};
    const float* ass[3] = {as1, as2, as3};
    const float* ads[3] = {ad1, ad2, ad3};
    const float* bs[3]  = {b1, b2, b3};
    const float* gs[3]  = {g1, g2, g3};
    const float* bes[3] = {be1, be2, be3};

    for (int L = 0; L < 3; ++L) {
        if (L == 0)
            gemm_wmma_bf16<64, false><<<mtiles, 256, 0, stream>>>(layer_in[L], Ws[L], nullptr, XL, N);
        else
            gemm_wmma_bf16<128, false><<<mtiles, 256, 0, stream>>>(layer_in[L], Ws[L], nullptr, XL, N);

        attn_coef_kernel<<<cdiv((long long)N * HEADS, 256), 256, 0, stream>>>(
            XL, ass[L], ads[L], SA, SD, N, HEADS, CHAN);

        hipMemsetAsync(Mx,  0, (size_t)N * HEADS * 4, stream);
        hipMemsetAsync(DEN, 0, (size_t)N * HEADS * 4, stream);
        hipMemsetAsync(ACC, 0, nf * 4, stream);

        edge_max_kernel<<<egrid, 256, 0, stream>>>(src, dst, SA, SD, Mx, E, N, HEADS);
        edge_expsum_kernel<<<egrid, 256, 0, stream>>>(src, dst, SA, SD, Mx, DEN, E, N, HEADS);
        edge_alpha_kernel<<<egrid, 256, 0, stream>>>(src, dst, SA, SD, Mx, DEN, ALPHA, E, N, HEADS);
        edge_accum128_kernel<<<cdiv((long long)ET * 32, 256), 256, 0, stream>>>(
            src, dst, ALPHA, XL, ACC, E, N);

        finalize_ln_kernel<<<cdiv(N, 8), 256, 0, stream>>>(
            ACC, bs[L], gs[L], bes[L], Hb, N);
    }

    // ---------------- output GAT layer (heads=1, C=2, mean) ----------------
    project_out_kernel<<<cdiv(N, 8), 256, 0, stream>>>(Hb, Wo, aso, ado, XL, SA, SD, N);

    hipMemsetAsync(Mx,  0, (size_t)N * 4, stream);
    hipMemsetAsync(DEN, 0, (size_t)N * 4, stream);
    hipMemsetAsync(ACC, 0, (size_t)N * 2 * 4, stream);

    edge_max_kernel<<<egrid, 256, 0, stream>>>(src, dst, SA, SD, Mx, E, N, 1);
    edge_expsum_kernel<<<egrid, 256, 0, stream>>>(src, dst, SA, SD, Mx, DEN, E, N, 1);
    edge_alpha_kernel<<<egrid, 256, 0, stream>>>(src, dst, SA, SD, Mx, DEN, ALPHA, E, N, 1);
    edge_accum2_kernel<<<egrid, 256, 0, stream>>>(src, dst, ALPHA, XL, ACC, E, N);

    finalize_out_kernel<<<cdiv((long long)N * 2, 256), 256, 0, stream>>>(
        ACC, bo, (float*)d_out, N);
}
